// LiTEN_40089224740933
// MI455X (gfx1250) — matrix operand order repaired and checked by reading, not compiled, not run on hardware
//
#include <hip/hip_runtime.h>
#include <hip/hip_bf16.h>
#include <math.h>

#define H      256
#define HEADS  8
#define HD     32
#define NRBF   32
#define NL     6
#define CUTR   5.0f
#define EPSF   1e-8f
#define PI_F   3.14159265358979323846f

typedef __bf16 bf16_t;
typedef bf16_t bf16x8  __attribute__((ext_vector_type(8)));
typedef bf16_t bf16x16 __attribute__((ext_vector_type(16)));
typedef float  f32x8   __attribute__((ext_vector_type(8)));

// ---------------- helpers ----------------

__device__ __forceinline__ bf16_t f2bf(float f) {
    unsigned u = __builtin_bit_cast(unsigned, f);
    unsigned r = u + 0x7FFFu + ((u >> 16) & 1u);   // round-to-nearest-even
    unsigned short s = (unsigned short)(r >> 16);
    return __builtin_bit_cast(bf16_t, s);
}

__device__ __forceinline__ float silu_f(float x) { return x / (1.0f + expf(-x)); }

__device__ __forceinline__ float cutoff_f(float d) {
    return (d < CUTR) ? 0.5f * (cosf(d * PI_F / CUTR) + 1.0f) : 0.0f;
}

__device__ __forceinline__ float wave_sum(float v) {
#pragma unroll
    for (int o = 16; o > 0; o >>= 1) v += __shfl_xor(v, o, 32);
    return v;
}

__device__ float block_sum(float v) {
    __shared__ float sm[8];
    v = wave_sum(v);
    int lane = threadIdx.x & 31, w = threadIdx.x >> 5, nw = blockDim.x >> 5;
    if (lane == 0) sm[w] = v;
    __syncthreads();
    float r = (threadIdx.x < (unsigned)nw) ? sm[threadIdx.x] : 0.0f;
    if (w == 0) r = wave_sum(r);
    if (threadIdx.x == 0) sm[0] = r;
    __syncthreads();
    r = sm[0];
    __syncthreads();
    return r;
}

// ---------------- WMMA GEMM ----------------
// C[MxNc] = act(A[MxK](bf16, row-major) * W + bias), W given pre-transposed
// as Wt[Nc x K] bf16 so each lane's B-fragment is a contiguous 32B chunk.
// One wave computes a 16x64 output strip: 4 accumulators, A-fragment reused 4x,
// 4 back-to-back v_wmma_f32_16x16x32_bf16 per K-step.

template <int ACT>
__global__ void gemm_wmma(const bf16_t* __restrict__ A,
                          const bf16_t* __restrict__ Wt,
                          const float* __restrict__ bias,
                          float* __restrict__ C,
                          int M, int K, int Nc) {
    int wid    = blockIdx.x * (blockDim.x >> 5) + (threadIdx.x >> 5);
    int ngrp   = Nc >> 6;                 // groups of 4 column-tiles (64 cols)
    int mtiles = (M + 15) >> 4;
    if (wid >= mtiles * ngrp) return;     // whole-wave exit: EXEC stays all-1s
    int mt = wid / ngrp, ng = wid - mt * ngrp;
    int lane = threadIdx.x & 31;
    int lrow = lane & 15;
    bool hi  = lane >= 16;
    int rowA = mt * 16 + lrow; if (rowA >= M) rowA = M - 1;
    // A frag: lanes 0-15 hold K k0..k0+15, hi lanes k0+16..k0+31.
    const bf16_t* Ar = A + (size_t)rowA * K + (hi ? 16 : 0);
    // B frag: lanes 0-15 hold K k0..k0+15, hi lanes k0+16..k0+31, col = lane&15.
    const bf16_t* Bc = Wt + (hi ? 16 : 0);
    int col0 = ng * 64 + lrow;

    f32x8 acc[4] = {};
    for (int k0 = 0; k0 < K; k0 += 32) {
        bf16x8 a0 = *(const bf16x8*)(Ar + k0);
        bf16x8 a1 = *(const bf16x8*)(Ar + k0 + 8);
        bf16x16 a = __builtin_shufflevector(a0, a1,
                        0, 1, 2, 3, 4, 5, 6, 7, 8, 9, 10, 11, 12, 13, 14, 15);
#pragma unroll
        for (int j = 0; j < 4; ++j) {
            bf16x16 b = *(const bf16x16*)(Bc + (size_t)(col0 + 16 * j) * K + k0);
            acc[j] = __builtin_amdgcn_wmma_f32_16x16x32_bf16(
                false, a, false, b, (short)0, acc[j], false, false);
        }
    }
#pragma unroll
    for (int j = 0; j < 4; ++j) {
        int col = col0 + 16 * j;
        float bv = bias ? bias[col] : 0.0f;
#pragma unroll
        for (int r = 0; r < 8; ++r) {
            int row = mt * 16 + r + (hi ? 8 : 0);
            if (row < M) {
                float val = acc[j][r] + bv;
                if (ACT == 1) val = silu_f(val);
                C[(size_t)row * Nc + col] = val;
            }
        }
    }
}

// ---------------- conversion kernels ----------------

// Wt[n*K + k] = bf16(W[k*Nc + n])  (transpose + convert, run once per weight)
__global__ void k_wtrans(const float* __restrict__ W, bf16_t* __restrict__ Wt,
                         int K, int Nc) {
    int i = blockIdx.x * blockDim.x + threadIdx.x;
    if (i >= K * Nc) return;
    int n = i / K, k = i - n * K;
    Wt[i] = f2bf(W[(size_t)k * Nc + n]);
}

__global__ void k_cvt_bf16(const float* __restrict__ in, bf16_t* __restrict__ out,
                           long n) {
    long i = (long)blockIdx.x * blockDim.x + threadIdx.x;
    if (i < n) out[i] = f2bf(in[i]);
}

// ---------------- elementwise / reduction kernels ----------------

__global__ void k_embed(const int* __restrict__ z, const float* __restrict__ emb,
                        float* __restrict__ x, int N) {
    int n = blockIdx.x, t = threadIdx.x;
    x[(size_t)n * H + t] = emb[(size_t)z[n] * H + t];
}

// per-edge: geometry + ExpNormal RBF + f0 = rbf @ We + be  (dual-write f32 + bf16)
__global__ void k_edge_init(const float* __restrict__ pos, const float* __restrict__ shift,
                            const int* __restrict__ src, const int* __restrict__ dst,
                            const float* __restrict__ eew, const float* __restrict__ eeb,
                            float* __restrict__ dist, float* __restrict__ dij,
                            float* __restrict__ f, bf16_t* __restrict__ fbf, int E) {
    int e = blockIdx.x, t = threadIdx.x;
    int s_ = src[e], d_ = dst[e];
    float vx = pos[s_ * 3 + 0] - pos[d_ * 3 + 0] + shift[(size_t)e * 3 + 0];
    float vy = pos[s_ * 3 + 1] - pos[d_ * 3 + 1] + shift[(size_t)e * 3 + 1];
    float vz = pos[s_ * 3 + 2] - pos[d_ * 3 + 2] + shift[(size_t)e * 3 + 2];
    float sq = vx * vx + vy * vy + vz * vz;
    float dd = sqrtf(fmaxf(sq, EPSF));
    float inv = (sq > EPSF) ? (1.0f / dd) : 0.0f;
    if (t == 0) {
        dist[e] = dd;
        dij[(size_t)e * 3 + 0] = vx * inv;
        dij[(size_t)e * 3 + 1] = vy * inv;
        dij[(size_t)e * 3 + 2] = vz * inv;
    }
    __shared__ float rbf[NRBF];
    if (t < NRBF) {
        float m0   = expf(-CUTR);
        float mean = m0 + (float)t * (1.0f - m0) / (float)(NRBF - 1);
        float tb   = (2.0f / (float)NRBF) * (1.0f - m0);
        float beta = 1.0f / (tb * tb);
        float g    = expf(-dd) - mean;           // alpha = 5/CUTOFF = 1
        rbf[t] = cutoff_f(dd) * expf(-beta * g * g);
    }
    __syncthreads();
    float acc = eeb[t];
#pragma unroll
    for (int r = 0; r < NRBF; ++r) acc += rbf[r] * eew[r * H + t];
    f[(size_t)e * H + t]   = acc;
    fbf[(size_t)e * H + t] = f2bf(acc);
}

// layernorm -> bf16 output (consumed only by GEMMs)
__global__ void k_layernorm(const float* __restrict__ x, const float* __restrict__ sc,
                            const float* __restrict__ bi, bf16_t* __restrict__ y, int N) {
    int n = blockIdx.x, t = threadIdx.x;
    float v   = x[(size_t)n * H + t];
    float mu  = block_sum(v) * (1.0f / H);
    float dv_ = v - mu;
    float var = block_sum(dv_ * dv_) * (1.0f / H);
    y[(size_t)n * H + t] = f2bf(dv_ * rsqrtf(var + 1e-5f) * sc[t] + bi[t]);
}

// vec RMS norm, dual-write f32 (for aggregation) + bf16 (for GEMMs)
__global__ void k_vecnorm(const float* __restrict__ vec, float* __restrict__ out,
                          bf16_t* __restrict__ outbf, int N) {
    int n = blockIdx.x, t = threadIdx.x;
    size_t b = (size_t)n * 3 * H;
    float a0 = vec[b + t], a1 = vec[b + H + t], a2 = vec[b + 2 * H + t];
    float mean = block_sum(a0 * a0 + a1 * a1 + a2 * a2) * (1.0f / H);
    float inv  = rsqrtf(mean + EPSF);
    out[b + t] = a0 * inv;  out[b + H + t] = a1 * inv;  out[b + 2 * H + t] = a2 * inv;
    outbf[b + t] = f2bf(a0 * inv);
    outbf[b + H + t] = f2bf(a1 * inv);
    outbf[b + 2 * H + t] = f2bf(a2 * inv);
}

// attn: one wave per head (HD == 32); vj dual-write f32 + bf16
__global__ void k_attn(const float* __restrict__ q, const float* __restrict__ k,
                       const float* __restrict__ v, const float* __restrict__ dk,
                       const float* __restrict__ dv, const float* __restrict__ dist,
                       const int* __restrict__ src, const int* __restrict__ dst,
                       float* __restrict__ vj, bf16_t* __restrict__ vjbf, int E) {
    int e = blockIdx.x, t = threadIdx.x;
    int s_ = src[e], d_ = dst[e];
    float p = q[(size_t)d_ * H + t] * k[(size_t)s_ * H + t] * dk[(size_t)e * H + t];
    float sum = wave_sum(p);
    float a = silu_f(sum) * cutoff_f(dist[e]);
    float r = v[(size_t)s_ * H + t] * dv[(size_t)e * H + t] * a;
    vj[(size_t)e * H + t]   = r;
    vjbf[(size_t)e * H + t] = f2bf(r);
}

// scatter-add of vj and vec_j into node accumulators
__global__ void k_agg(const float* __restrict__ vj, const float* __restrict__ s12,
                      const float* __restrict__ vn, const float* __restrict__ dij,
                      const int* __restrict__ src, const int* __restrict__ dst,
                      float* __restrict__ xagg, float* __restrict__ vecagg, int E) {
    int e = blockIdx.x, t = threadIdx.x;
    int s_ = src[e], d_ = dst[e];
    float vje = vj[(size_t)e * H + t];
    atomicAdd(&xagg[(size_t)d_ * H + t], vje);
    float s1 = s12[(size_t)e * 2 * H + t];
    float s2 = s12[(size_t)e * 2 * H + H + t];
#pragma unroll
    for (int dim = 0; dim < 3; ++dim) {
        float vc = vn[(size_t)s_ * 3 * H + dim * H + t] * s1 + s2 * dij[(size_t)e * 3 + dim];
        atomicAdd(&vecagg[(size_t)d_ * 3 * H + dim * H + t], vc);
    }
}

__global__ void k_node_update(float* __restrict__ x, float* __restrict__ vec,
                              const float* __restrict__ vecp, const float* __restrict__ o,
                              const float* __restrict__ vecagg, int N) {
    int n = blockIdx.x, t = threadIdx.x;
    const float* vp = vecp + (size_t)n * 3 * (3 * H);
    float vdot = 0.0f;
#pragma unroll
    for (int d = 0; d < 3; ++d)
        vdot += vp[d * (3 * H) + t] * vp[d * (3 * H) + H + t];
    const float* on = o + (size_t)n * (3 * H);
    float o1 = on[t], o2 = on[H + t], o3 = on[2 * H + t];
    x[(size_t)n * H + t] += vdot * o2 + o3;
#pragma unroll
    for (int d = 0; d < 3; ++d) {
        size_t idx = (size_t)n * 3 * H + d * H + t;
        vec[idx] += vp[d * (3 * H) + 2 * H + t] * o1 + vecagg[idx];
    }
}

// f += g * sum_d reject(wt,d)*reject(ws,d); dual-write f32 + bf16
__global__ void k_edge_update(float* __restrict__ f, bf16_t* __restrict__ fbf,
                              const float* __restrict__ g,
                              const float* __restrict__ wtf, const float* __restrict__ wsf,
                              const float* __restrict__ dij,
                              const int* __restrict__ src, const int* __restrict__ dst, int E) {
    int e = blockIdx.x, t = threadIdx.x;
    int s_ = src[e], d_ = dst[e];
    float dx = dij[(size_t)e * 3 + 0], dy = dij[(size_t)e * 3 + 1], dz = dij[(size_t)e * 3 + 2];
    size_t bt = (size_t)d_ * 3 * H, bs = (size_t)s_ * 3 * H;
    float wt0 = wtf[bt + t], wt1 = wtf[bt + H + t], wt2 = wtf[bt + 2 * H + t];
    float ws0 = wsf[bs + t], ws1 = wsf[bs + H + t], ws2 = wsf[bs + 2 * H + t];
    float pt = wt0 * dx + wt1 * dy + wt2 * dz;
    wt0 -= pt * dx; wt1 -= pt * dy; wt2 -= pt * dz;
    float ps = ws0 * dx + ws1 * dy + ws2 * dz;
    ws0 -= ps * dx; ws1 -= ps * dy; ws2 -= ps * dz;
    float dot = wt0 * ws0 + wt1 * ws1 + wt2 * ws2;
    size_t i = (size_t)e * H + t;
    float nf = f[i] + g[i] * dot;
    f[i]   = nf;
    fbf[i] = f2bf(nf);
}

__global__ void k_readout(const float* __restrict__ hid, const float* __restrict__ w2,
                          const float* __restrict__ b2, const int* __restrict__ batch,
                          float* __restrict__ out, int N) {
    int n = blockIdx.x, t = threadIdx.x;     // blockDim = H/2 = 128
    float p = hid[(size_t)n * (H / 2) + t] * w2[t];
    float s = block_sum(p);
    if (t == 0) atomicAdd(&out[batch[n]], s + b2[0]);
}

// ---------------- host launch ----------------

extern "C" void kernel_launch(void* const* d_in, const int* in_sizes, int n_in,
                              void* d_out, int out_size, void* d_ws, size_t ws_size,
                              hipStream_t stream) {
    (void)n_in; (void)ws_size;
    int idx = 0;
    const int*   z     = (const int*)d_in[idx++];   int N = in_sizes[0];
    const float* pos   = (const float*)d_in[idx++];
    const int*   batch = (const int*)d_in[idx++];
    const int*   ei    = (const int*)d_in[idx++];   int E = in_sizes[3] / 2;
    idx++; /* n_graph scalar (unused: out_size gives G) */
    const float* shift = (const float*)d_in[idx++];
    const float* emb   = (const float*)d_in[idx++];
    const float* eew   = (const float*)d_in[idx++];
    const float* eeb   = (const float*)d_in[idx++];
    const float* onS   = (const float*)d_in[idx++];
    const float* onB   = (const float*)d_in[idx++];
    const float* r1w   = (const float*)d_in[idx++];
    const float* r1b   = (const float*)d_in[idx++];
    const float* r2w   = (const float*)d_in[idx++];
    const float* r2b   = (const float*)d_in[idx++];

    struct LP {
        const float *lnS,*lnB,*qw,*qb,*kw,*kb,*vw,*vb,*dkw,*dkb,*dvw,*dvb,
                    *vecw,*sw,*sb,*ow,*ob,*wsw,*wtw,*fw,*fb;
    } lp[NL];
    for (int L = 0; L < NL; ++L) {
        lp[L].lnS  = (const float*)d_in[idx++]; lp[L].lnB = (const float*)d_in[idx++];
        lp[L].qw   = (const float*)d_in[idx++]; lp[L].qb  = (const float*)d_in[idx++];
        lp[L].kw   = (const float*)d_in[idx++]; lp[L].kb  = (const float*)d_in[idx++];
        lp[L].vw   = (const float*)d_in[idx++]; lp[L].vb  = (const float*)d_in[idx++];
        lp[L].dkw  = (const float*)d_in[idx++]; lp[L].dkb = (const float*)d_in[idx++];
        lp[L].dvw  = (const float*)d_in[idx++]; lp[L].dvb = (const float*)d_in[idx++];
        lp[L].vecw = (const float*)d_in[idx++];
        lp[L].sw   = (const float*)d_in[idx++]; lp[L].sb  = (const float*)d_in[idx++];
        lp[L].ow   = (const float*)d_in[idx++]; lp[L].ob  = (const float*)d_in[idx++];
        lp[L].wsw  = (const float*)d_in[idx++];
        lp[L].wtw  = (const float*)d_in[idx++];
        if (L < NL - 1) { lp[L].fw = (const float*)d_in[idx++]; lp[L].fb = (const float*)d_in[idx++]; }
        else            { lp[L].fw = nullptr;                   lp[L].fb = nullptr; }
    }
    const int* src = ei;
    const int* dst = ei + E;

    // ---- workspace carve ----
    char* base = (char*)d_ws;
    size_t off = 0;
    auto carve = [&](size_t elems) {                 // f32
        float* p = (float*)(base + off);
        off += ((elems * sizeof(float) + 255) & ~(size_t)255);
        return p;
    };
    auto carve_bf = [&](size_t elems) {              // bf16
        bf16_t* p = (bf16_t*)(base + off);
        off += ((elems * sizeof(bf16_t) + 255) & ~(size_t)255);
        return p;
    };
    float* dist   = carve(E);
    float* dij    = carve((size_t)E * 3);
    float* fbuf   = carve((size_t)E * H);
    float* s12    = carve((size_t)E * 2 * H);   // dk/dv live here first, then s12 reuses it
    float* dk     = s12;
    float* dv     = s12 + (size_t)E * H;
    float* vj     = carve((size_t)E * H);       // later reused for g = silu(f@Wf)
    float* x      = carve((size_t)N * H);
    float* q      = carve((size_t)N * H);
    float* kk     = carve((size_t)N * H);
    float* vv     = carve((size_t)N * H);
    float* vec    = carve((size_t)N * 3 * H);
    float* vecn   = carve((size_t)N * 3 * H);
    float* vecp   = carve((size_t)N * 3 * 3 * H);
    float* xagg   = carve((size_t)N * H);
    float* vecagg = carve((size_t)N * 3 * H);
    float* obuf   = carve((size_t)N * 3 * H);
    float* wtf    = carve((size_t)N * 3 * H);
    float* wsf    = carve((size_t)N * 3 * H);
    float* hid    = carve((size_t)N * (H / 2));
    // bf16 activation mirrors (GEMM A-operands)
    bf16_t* f_bf    = carve_bf((size_t)E * H);
    bf16_t* vj_bf   = carve_bf((size_t)E * H);
    bf16_t* xn_bf   = carve_bf((size_t)N * H);
    bf16_t* vn_bf   = carve_bf((size_t)N * 3 * H);
    bf16_t* xagg_bf = carve_bf((size_t)N * H);

    // ---- weight pre-transpose+convert to bf16 [Nc x K] ----
    auto wconv = [&](const float* W, int K, int Nc) -> const bf16_t* {
        bf16_t* p = carve_bf((size_t)K * Nc);
        int total = K * Nc;
        k_wtrans<<<(total + 255) / 256, 256, 0, stream>>>(W, p, K, Nc);
        return p;
    };
    struct LPT { const bf16_t *qw,*kw,*vw,*dkw,*dvw,*vecw,*sw,*ow,*wsw,*wtw,*fw; } lt[NL];
    for (int L = 0; L < NL; ++L) {
        lt[L].qw   = wconv(lp[L].qw,   H, H);
        lt[L].kw   = wconv(lp[L].kw,   H, H);
        lt[L].vw   = wconv(lp[L].vw,   H, H);
        lt[L].dkw  = wconv(lp[L].dkw,  H, H);
        lt[L].dvw  = wconv(lp[L].dvw,  H, H);
        lt[L].vecw = wconv(lp[L].vecw, H, 3 * H);
        lt[L].sw   = wconv(lp[L].sw,   H, 2 * H);
        lt[L].ow   = wconv(lp[L].ow,   H, 3 * H);
        lt[L].wsw  = wconv(lp[L].wsw,  H, H);
        lt[L].wtw  = wconv(lp[L].wtw,  H, H);
        lt[L].fw   = (L < NL - 1) ? wconv(lp[L].fw, H, H) : nullptr;
    }
    const bf16_t* r1wt = wconv(r1w, H, H / 2);

    auto gemmN = [&](const bf16_t* A, const bf16_t* Wt, const float* b, float* C,
                     int M, int K, int Nc, int act) {
        int tiles  = ((M + 15) / 16) * (Nc / 64);   // one 16x64 strip per wave
        int blocks = (tiles + 7) / 8;               // 8 waves per 256-thread block
        if (act) gemm_wmma<1><<<blocks, 256, 0, stream>>>(A, Wt, b, C, M, K, Nc);
        else     gemm_wmma<0><<<blocks, 256, 0, stream>>>(A, Wt, b, C, M, K, Nc);
    };

    // ---- forward pass ----
    k_embed<<<N, H, 0, stream>>>(z, emb, x, N);
    hipMemsetAsync(vec, 0, (size_t)N * 3 * H * sizeof(float), stream);
    hipMemsetAsync(vn_bf, 0, (size_t)N * 3 * H * sizeof(bf16_t), stream);  // layer-0 vn
    k_edge_init<<<E, H, 0, stream>>>(pos, shift, src, dst, eew, eeb,
                                     dist, dij, fbuf, f_bf, E);

    for (int L = 0; L < NL; ++L) {
        k_layernorm<<<N, H, 0, stream>>>(x, lp[L].lnS, lp[L].lnB, xn_bf, N);
        const float* vn = vec;                       // layer 0: zeros (vn_bf zeroed above)
        if (L > 0) { k_vecnorm<<<N, H, 0, stream>>>(vec, vecn, vn_bf, N); vn = vecn; }

        gemmN(xn_bf, lt[L].qw, lp[L].qb, q,  N, H, H, 0);
        gemmN(xn_bf, lt[L].kw, lp[L].kb, kk, N, H, H, 0);
        gemmN(xn_bf, lt[L].vw, lp[L].vb, vv, N, H, H, 0);
        gemmN(f_bf, lt[L].dkw, lp[L].dkb, dk, E, H, H, 1);
        gemmN(f_bf, lt[L].dvw, lp[L].dvb, dv, E, H, H, 1);

        k_attn<<<E, H, 0, stream>>>(q, kk, vv, dk, dv, dist, src, dst, vj, vj_bf, E);

        gemmN(vj_bf, lt[L].sw, lp[L].sb, s12, E, H, 2 * H, 1);   // dk/dv dead -> reuse

        hipMemsetAsync(xagg,   0, (size_t)N * H * sizeof(float), stream);
        hipMemsetAsync(vecagg, 0, (size_t)N * 3 * H * sizeof(float), stream);
        k_agg<<<E, H, 0, stream>>>(vj, s12, vn, dij, src, dst, xagg, vecagg, E);
        k_cvt_bf16<<<((long)N * H + 255) / 256, 256, 0, stream>>>(xagg, xagg_bf, (long)N * H);

        gemmN(vn_bf,   lt[L].vecw, nullptr,  vecp, 3 * N, H, 3 * H, 0);
        gemmN(xagg_bf, lt[L].ow,   lp[L].ob, obuf, N, H, 3 * H, 0);

        if (L < NL - 1) {
            gemmN(vn_bf, lt[L].wtw, nullptr, wtf, 3 * N, H, H, 0);
            gemmN(vn_bf, lt[L].wsw, nullptr, wsf, 3 * N, H, H, 0);
            gemmN(f_bf,  lt[L].fw,  lp[L].fb, vj, E, H, H, 1);   // g reuses dead vj buffer
        }

        k_node_update<<<N, H, 0, stream>>>(x, vec, vecp, obuf, vecagg, N);

        if (L < NL - 1)
            k_edge_update<<<E, H, 0, stream>>>(fbuf, f_bf, vj, wtf, wsf, dij, src, dst, E);
    }

    k_layernorm<<<N, H, 0, stream>>>(x, onS, onB, xn_bf, N);
    gemmN(xn_bf, r1wt, r1b, hid, N, H, H / 2, 1);
    hipMemsetAsync(d_out, 0, (size_t)out_size * sizeof(float), stream);
    k_readout<<<N, H / 2, 0, stream>>>(hid, r2w, r2b, batch, (float*)d_out, N);
}
